// SpecFormer_11948599017836
// MI455X (gfx1250) — matrix operand order repaired and checked by reading, not compiled, not run on hardware
//
#include <hip/hip_runtime.h>
#include <math.h>

// ---------------- sizes ----------------
#define Bn   2
#define Nn   256
#define Dn   128
#define INn  128
#define Hn   8
#define ROWS (Bn*Nn*Nn)      // 131072 flattened (b,n,m) rows
#define NND  (ROWS*Dn)       // 16777216 elements of [B,N,N,D]

typedef __attribute__((ext_vector_type(16))) __bf16 v16bf;
typedef __attribute__((ext_vector_type(8)))  float  v8f;

// ---------------------------------------------------------------------------
// WMMA 16x16 tile GEMM core: C(16x16,f32) += A(16x32,bf16) x B(32x16,bf16)
// A: row-major [M,K] bf16;  Bt: row-major [N,K] bf16 (i.e. B transposed)
// Fragment layouts follow CDNA5 ISA 7.12.2 (wave32).
// ---------------------------------------------------------------------------
__device__ __forceinline__ v8f wmma_tile(const __bf16* __restrict__ A, int lda,
                                         const __bf16* __restrict__ Bt, int ldb,
                                         int K, int m0, int n0, int lane) {
  v8f acc = {0.f,0.f,0.f,0.f,0.f,0.f,0.f,0.f};
  const int mr    = m0 + (lane & 15);
  const int nr    = n0 + (lane & 15);
  const int khalf = (lane >> 4) * 8;    // A: K sub-block for this half-wave
  const int kb    = (lane >> 4) * 8;    // B: K sub-block for this half-wave
  for (int kk = 0; kk < K; kk += 32) {
    v16bf a, b;
    const __bf16* pa0 = A + (long)mr * lda + kk + khalf;       // K = kk+khalf..+7
    const __bf16* pa1 = pa0 + 16;                              // K = kk+16+khalf..+7
#pragma unroll
    for (int j = 0; j < 8; ++j) { a[j] = pa0[j]; a[8 + j] = pa1[j]; }
    const __bf16* pb = Bt + (long)nr * ldb + kk + kb;          // K = kk+kb..+7, kk+16+kb..+7
#pragma unroll
    for (int j = 0; j < 8; ++j) { b[j] = pb[j]; b[8 + j] = pb[16 + j]; }
    acc = __builtin_amdgcn_wmma_f32_16x16x32_bf16(
        /*neg_a=*/false, a, /*neg_b=*/false, b,
        /*c_mod=*/(short)0, acc, /*reuse_a=*/false, /*reuse_b=*/false);
  }
  return acc;
}

// C/D layout: lane column = n0+(lane&15); VGPR v row = m0 + v + (lane>>4)*8.

// ---------------------------------------------------------------------------
// small helpers
// ---------------------------------------------------------------------------
__device__ __forceinline__ float gelu_exact(float x) {
  return 0.5f * x * (1.f + erff(x * 0.70710678118654752f));
}

// block-wide LayerNorm over 128 channels (128 threads, red[] shared scratch)
__device__ __forceinline__ float ln_block(float x, int c, float* red,
                                          const float* g, const float* b) {
  red[c] = x; __syncthreads();
  for (int s = 64; s > 0; s >>= 1) { if (c < s) red[c] += red[c + s]; __syncthreads(); }
  float mean = red[0] * (1.f / 128.f); __syncthreads();
  float d = x - mean;
  red[c] = d * d; __syncthreads();
  for (int s = 64; s > 0; s >>= 1) { if (c < s) red[c] += red[c + s]; __syncthreads(); }
  float var = red[0] * (1.f / 128.f); __syncthreads();
  return d * rsqrtf(var + 1e-5f) * g[c] + b[c];
}

// ---------------------------------------------------------------------------
// 1) weight prep: transpose fp32 [K,N] weights -> bf16 [N,K]
// ---------------------------------------------------------------------------
__global__ void k_prep_w(const float* fe_w1, const float* fe_w2,
                         const float* fenc_w2, const float* pre_w,
                         __bf16* fe_w1t, __bf16* fe_w2t,
                         __bf16* fenc_w2t, __bf16* preWt) {
  int t = blockIdx.x * 256 + threadIdx.x;
  if (t < 256 * 128) {                 // fe_w1: K=256,N=128 -> [128,256]
    int c = t / 256, k = t % 256;
    fe_w1t[t] = (__bf16)fe_w1[k * 128 + c];
  }
  if (t < 128 * 128) {                 // 128x128 ones
    int c = t / 128, k = t % 128;
    fe_w2t[t]   = (__bf16)fe_w2[k * 128 + c];
    fenc_w2t[t] = (__bf16)fenc_w2[k * 128 + c];
    preWt[t]    = (__bf16)pre_w[k * 128 + c];
  }
}

// 2) z = concat(node_fts, hidden) -> bf16 [B*N, 2*IN]
__global__ void k_concat(const float* node, const float* hidden, __bf16* zbf) {
  int t = blockIdx.x * 256 + threadIdx.x;          // 0..131071
  if (t >= Bn * Nn * 2 * INn) return;
  int c = t & 255, r = t >> 8;
  float v = (c < INn) ? node[r * INn + c] : hidden[r * INn + (c - INn)];
  zbf[t] = (__bf16)v;
}

// 3) combinatorial Laplacian
__global__ void k_lap(const float* adj, float* lap) {
  int t = blockIdx.x * 256 + threadIdx.x;
  if (t >= Bn * Nn) return;
  const float* row = adj + (long)t * Nn;
  float deg = 0.f;
  for (int j = 0; j < Nn; ++j) deg += row[j];
  float* lrow = lap + (long)t * Nn;
  int i = t & (Nn - 1);
  for (int j = 0; j < Nn; ++j) lrow[j] = (i == j ? deg : 0.f) - row[j];
}

// 4) parallel cyclic Jacobi eigensolver, A in LDS (256 KB of the 320 KB WGP LDS)
__global__ void k_jacobi(const float* lap, float* V, float* evals) {
  extern __shared__ float A[];                 // 256*256 floats
  __shared__ int   topI[128], botI[128];
  __shared__ float cS[128], sS[128];
  const int b = blockIdx.x, tid = threadIdx.x;
  const float* src = lap + (long)b * Nn * Nn;
  float* Vb = V + (long)b * Nn * Nn;
  for (int i = tid; i < Nn * Nn; i += 256) {
    A[i] = src[i];
    Vb[i] = ((i >> 8) == (i & 255)) ? 1.f : 0.f;
  }
  if (tid < 128) { topI[tid] = 2 * tid; botI[tid] = 2 * tid + 1; }
  __syncthreads();
  for (int sweep = 0; sweep < 6; ++sweep) {
    for (int round = 0; round < Nn - 1; ++round) {
      if (tid < 128) {
        int p = topI[tid], q = botI[tid];
        float app = A[p * Nn + p], aqq = A[q * Nn + q], apq = A[p * Nn + q];
        float c = 1.f, s = 0.f;
        if (fabsf(apq) > 1e-12f) {
          float th = (aqq - app) / (2.f * apq);
          float tt = 1.f / (fabsf(th) + sqrtf(1.f + th * th));
          if (th < 0.f) tt = -tt;
          c = rsqrtf(1.f + tt * tt);
          s = tt * c;
        }
        cS[tid] = c; sS[tid] = s;
      }
      __syncthreads();
      for (int w = tid; w < 128 * Nn; w += 256) {      // column rotations
        int k = w >> 8, i = w & 255;
        int p = topI[k], q = botI[k];
        float c = cS[k], s = sS[k];
        float x = A[i * Nn + p], y = A[i * Nn + q];
        A[i * Nn + p] = c * x - s * y;
        A[i * Nn + q] = s * x + c * y;
      }
      __syncthreads();
      for (int w = tid; w < 128 * Nn; w += 256) {      // row rotations
        int k = w >> 8, i = w & 255;
        int p = topI[k], q = botI[k];
        float c = cS[k], s = sS[k];
        float x = A[p * Nn + i], y = A[q * Nn + i];
        A[p * Nn + i] = c * x - s * y;
        A[q * Nn + i] = s * x + c * y;
      }
      __syncthreads();
      for (int w = tid; w < 128 * Nn; w += 256) {      // eigenvector columns
        int k = w >> 8, i = w & 255;
        int p = topI[k], q = botI[k];
        float c = cS[k], s = sS[k];
        float x = Vb[i * Nn + p], y = Vb[i * Nn + q];
        Vb[i * Nn + p] = c * x - s * y;
        Vb[i * Nn + q] = s * x + c * y;
      }
      __syncthreads();
      if (tid == 0) {                                   // round-robin advance
        int lastTop = topI[127], firstBot = botI[0];
        for (int i = 127; i >= 2; --i) topI[i] = topI[i - 1];
        for (int i = 0; i < 127; ++i) botI[i] = botI[i + 1];
        topI[1] = firstBot; botI[127] = lastTop;
      }
      __syncthreads();
    }
  }
  for (int i = tid; i < Nn; i += 256) evals[b * Nn + i] = A[i * Nn + i];
}

// 5) spectral prep: Ebf = bf16(V);  Abf[b,h] = bf16(V[:,k] * (eval_k*dec_w[h]+dec_b[h]))
__global__ void k_specprep(const float* V, const float* evals, const float* dec_w,
                           const float* dec_b, __bf16* Abf, __bf16* Ebf) {
  int t = blockIdx.x * 256 + threadIdx.x;          // over B*N*N
  if (t >= Bn * Nn * Nn) return;
  int k = t & 255, n = (t >> 8) & 255, b = t >> 16;
  float v = V[t];
  Ebf[t] = (__bf16)v;
  float ev = evals[b * Nn + k];
#pragma unroll
  for (int h = 0; h < Hn; ++h) {
    float ne = ev * dec_w[h] + dec_b[h];
    Abf[((long)(b * Hn + h) * Nn * Nn) + n * Nn + k] = (__bf16)(v * ne);
  }
}

// 6) generic f32->bf16 convert
__global__ void k_cvt(const float* src, __bf16* dst, long n) {
  long t = (long)blockIdx.x * 256 + threadIdx.x;
  if (t < n) dst[t] = (__bf16)src[t];
}

// 7) generic WMMA GEMM: C[M,N] = act(A[M,K] x Bt[N,K]^T + bias)
__global__ void k_gemm(const __bf16* A, const __bf16* Bt, const float* bias,
                       float* C, int M, int N, int K, int relu) {
  int lane = threadIdx.x & 31;
  int wave = blockIdx.x * (blockDim.x >> 5) + (threadIdx.x >> 5);
  int tilesN = N >> 4;
  int tm = wave / tilesN, tn = wave % tilesN;
  if (tm * 16 >= M) return;                        // uniform across the wave
  v8f acc = wmma_tile(A, K, Bt, K, K, tm * 16, tn * 16, lane);
  int col = tn * 16 + (lane & 15);
  float bv = bias ? bias[col] : 0.f;
#pragma unroll
  for (int v = 0; v < 8; ++v) {
    int row = tm * 16 + v + ((lane >> 4) << 3);
    float x = acc[v] + bv;
    if (relu) x = fmaxf(x, 0.f);
    C[(long)row * N + col] = x;
  }
}

// 8) filters[b,:,:,h] = (V*ne_h) @ V^T   (one 16x16 tile per wave, WMMA)
__global__ void k_filters(const __bf16* Abf, const __bf16* Ebf, float* filt) {
  int lane = threadIdx.x & 31;
  int wave = blockIdx.x * 8 + (threadIdx.x >> 5);  // 4096 waves total
  int tile = wave & 255, bh = wave >> 8;           // bh in [0,16)
  int b = bh >> 3, h = bh & 7;
  int tm = tile >> 4, tn = tile & 15;
  const __bf16* Ap = Abf + (long)bh * Nn * Nn;
  const __bf16* Bp = Ebf + (long)b * Nn * Nn;      // Bt[m,k] = V[m,k]
  v8f acc = wmma_tile(Ap, Nn, Bp, Nn, Nn, tm * 16, tn * 16, lane);
  int mcol = tn * 16 + (lane & 15);
#pragma unroll
  for (int v = 0; v < 8; ++v) {
    int nrow = tm * 16 + v + ((lane >> 4) << 3);
    filt[(((long)(b * Nn + nrow) * Nn + mcol) * Hn) + h] = acc[v];
  }
}

// 9) bases stage 1: t1 = gelu(LN(bases @ fenc_w1 + b1)) -> bf16  (9->128 expand)
__global__ void k_bases1(const float* filt, const float* w1, const float* b1,
                         const float* ln_g, const float* ln_b, __bf16* t1g) {
  __shared__ float red[128];
  int r = blockIdx.x;                  // flattened (b,n,m)
  int c = threadIdx.x;                 // 0..127
  int m = r & 255, n = (r >> 8) & 255;
  const float* f = filt + (long)r * Hn;
  float x = b1[c] + (n == m ? w1[c] : 0.f);
#pragma unroll
  for (int h = 0; h < Hn; ++h) x += f[h] * w1[(h + 1) * Dn + c];
  float y = ln_block(x, c, red, ln_g, ln_b);
  t1g[(long)r * Dn + c] = (__bf16)gelu_exact(y);
}

// 10) bases stage 2: WMMA GEMM (131072x128x128) + LN + gelu + softmax + *adj
//     one workgroup per 16 rows; 8 waves cover the 8 column tiles.
__global__ void k_bases2(const __bf16* t1g, const __bf16* w2t, const float* b2,
                         const float* ln_g, const float* ln_b, const float* adj,
                         float* basesF) {
  __shared__ float tile[16 * Dn];
  __shared__ float red[16][16];
  __shared__ float stat[16];
  int lane = threadIdx.x & 31, wv = threadIdx.x >> 5;
  int m0 = blockIdx.x * 16;
  v8f acc = wmma_tile(t1g, Dn, w2t, Dn, Dn, m0, wv * 16, lane);
  int col = wv * 16 + (lane & 15);
#pragma unroll
  for (int v = 0; v < 8; ++v) {
    int row = v + ((lane >> 4) << 3);
    tile[row * Dn + col] = acc[v] + b2[col];
  }
  __syncthreads();
  int r = threadIdx.x >> 4, t = threadIdx.x & 15;  // 16 threads per row
  float* rowp = &tile[r * Dn];
  float s = 0.f;
  for (int j = t; j < Dn; j += 16) s += rowp[j];
  red[r][t] = s; __syncthreads();
  if (t == 0) { float a = 0; for (int j = 0; j < 16; ++j) a += red[r][j]; stat[r] = a * (1.f / Dn); }
  __syncthreads();
  float mean = stat[r];
  float v2 = 0.f;
  for (int j = t; j < Dn; j += 16) { float d = rowp[j] - mean; v2 += d * d; }
  red[r][t] = v2; __syncthreads();
  if (t == 0) { float a = 0; for (int j = 0; j < 16; ++j) a += red[r][j]; stat[r] = rsqrtf(a * (1.f / Dn) + 1e-5f); }
  __syncthreads();
  float inv = stat[r];
  for (int j = t; j < Dn; j += 16) {
    float x = (rowp[j] - mean) * inv * ln_g[j] + ln_b[j];
    rowp[j] = gelu_exact(x);
  }
  __syncthreads();
  float mx = -1e30f;
  for (int j = t; j < Dn; j += 16) mx = fmaxf(mx, rowp[j]);
  red[r][t] = mx; __syncthreads();
  if (t == 0) { float a = -1e30f; for (int j = 0; j < 16; ++j) a = fmaxf(a, red[r][j]); stat[r] = a; }
  __syncthreads();
  mx = stat[r];
  float es = 0.f;
  for (int j = t; j < Dn; j += 16) es += __expf(rowp[j] - mx);
  red[r][t] = es; __syncthreads();
  if (t == 0) { float a = 0; for (int j = 0; j < 16; ++j) a += red[r][j]; stat[r] = a; }
  __syncthreads();
  float denom = stat[r];
  int rg = m0 + r;
  int mm = rg & 255, nn2 = (rg >> 8) & 255, bb = rg >> 16;
  float aval = adj[((long)bb * Nn + nn2) * Nn + mm];
  for (int j = t; j < Dn; j += 16)
    basesF[(long)rg * Dn + j] = aval * __expf(rowp[j] - mx) / denom;
}

// 11) conv: e = gelu(edge @ pre_w + pre_b) * bases  (WMMA, writes e output)
__global__ void k_conv(const __bf16* edgebf, const __bf16* preWt, const float* pre_b,
                       const float* basesF, float* e_out) {
  int lane = threadIdx.x & 31;
  int wave = blockIdx.x * 8 + (threadIdx.x >> 5);  // 65536 waves
  int tm = wave >> 3, tn = wave & 7;
  v8f acc = wmma_tile(edgebf, Dn, preWt, Dn, Dn, tm * 16, tn * 16, lane);
  int col = tn * 16 + (lane & 15);
  float bv = pre_b[col];
#pragma unroll
  for (int v = 0; v < 8; ++v) {
    int row = tm * 16 + v + ((lane >> 4) << 3);
    long idx = (long)row * Dn + col;
    float x = gelu_exact(acc[v] + bv);
    e_out[idx] = x * basesF[idx];
  }
}

// 12) hbuf[b,m,c] = h0[b,m,c] + sum_n e[b,n,m,c]
__global__ void k_esum(const float* e, const float* h0, float* hbuf) {
  int t = blockIdx.x * 256 + threadIdx.x;          // 65536
  if (t >= Bn * Nn * Dn) return;
  int c = t & 127, m = (t >> 7) & 255, b = t >> 15;
  float s = h0[t];
  const float* base = e + ((long)b * Nn * Nn + m) * Dn + c;
  for (int n = 0; n < Nn; ++n) s += base[(long)n * Nn * Dn];
  hbuf[t] = s;
}

// 13) final residual MLP with LayerNorms -> h output
__global__ void k_final(const float* hbuf, const float* w1, const float* b1,
                        const float* g1, const float* bb1, const float* w2,
                        const float* b2, const float* g2, const float* bb2,
                        float* h_out) {
  __shared__ float hrow[128];
  __shared__ float buf[128];
  __shared__ float red[128];
  int r = blockIdx.x, c = threadIdx.x;
  float hv = hbuf[(long)r * Dn + c];
  hrow[c] = hv; __syncthreads();
  float x = b1[c];
  for (int k = 0; k < Dn; ++k) x += hrow[k] * w1[k * Dn + c];
  x = fmaxf(ln_block(x, c, red, g1, bb1), 0.f);
  buf[c] = x; __syncthreads();
  float z = b2[c];
  for (int k = 0; k < Dn; ++k) z += buf[k] * w2[k * Dn + c];
  z = fmaxf(ln_block(z, c, red, g2, bb2), 0.f);
  h_out[(long)r * Dn + c] = hv + z;
}

// ---------------------------------------------------------------------------
// host side
// ---------------------------------------------------------------------------
static inline char* carve(char*& p, size_t bytes) {
  char* r = p;
  p += (bytes + 255) & ~(size_t)255;
  return r;
}

extern "C" void kernel_launch(void* const* d_in, const int* in_sizes, int n_in,
                              void* d_out, int out_size, void* d_ws, size_t ws_size,
                              hipStream_t stream) {
  // inputs: 0 node_fts, 1 edge_fts, 2 graph_fts(unused), 3 hidden, 4 adj_mat, 5.. params
  const float* node   = (const float*)d_in[0];
  const float* edge   = (const float*)d_in[1];
  const float* hidden = (const float*)d_in[3];
  const float* adj    = (const float*)d_in[4];
#define PRM(i) ((const float*)d_in[5 + (i)])
  const float* fe_w1 = PRM(0),  *fe_b1 = PRM(1),  *fe_w2 = PRM(2),  *fe_b2 = PRM(3);
  const float* dec_w = PRM(18), *dec_b = PRM(19);
  const float* fenc_w1 = PRM(20), *fenc_b1 = PRM(21);
  const float* fenc_ln1_g = PRM(22), *fenc_ln1_b = PRM(23);
  const float* fenc_w2 = PRM(24), *fenc_b2 = PRM(25);
  const float* fenc_ln2_g = PRM(26), *fenc_ln2_b = PRM(27);
  const float* c0_pre_w = PRM(28), *c0_pre_b = PRM(29);
  const float* c0_w1 = PRM(30), *c0_b1 = PRM(31), *c0_ln1_g = PRM(32), *c0_ln1_b = PRM(33);
  const float* c0_w2 = PRM(34), *c0_b2 = PRM(35), *c0_ln2_g = PRM(36), *c0_ln2_b = PRM(37);

  float* h_out = (float*)d_out;                       // [B,N,D]  = 65536
  float* e_out = (float*)d_out + Bn * Nn * Dn;        // [B,N,N,D] = 16777216

  char* p = (char*)d_ws;
  float*  lap    = (float*) carve(p, (size_t)Bn*Nn*Nn*4);
  float*  V      = (float*) carve(p, (size_t)Bn*Nn*Nn*4);
  float*  evals  = (float*) carve(p, (size_t)Bn*Nn*4);
  __bf16* Abf    = (__bf16*)carve(p, (size_t)Bn*Hn*Nn*Nn*2);
  __bf16* Ebf    = (__bf16*)carve(p, (size_t)Bn*Nn*Nn*2);
  float*  filt   = (float*) carve(p, (size_t)ROWS*Hn*4);
  __bf16* zbf    = (__bf16*)carve(p, (size_t)Bn*Nn*2*INn*2);
  float*  u      = (float*) carve(p, (size_t)Bn*Nn*Dn*4);
  __bf16* ubf    = (__bf16*)carve(p, (size_t)Bn*Nn*Dn*2);
  float*  h0     = (float*) carve(p, (size_t)Bn*Nn*Dn*4);
  float*  hbuf   = (float*) carve(p, (size_t)Bn*Nn*Dn*4);
  __bf16* fe_w1t = (__bf16*)carve(p, (size_t)256*128*2);
  __bf16* fe_w2t = (__bf16*)carve(p, (size_t)128*128*2);
  __bf16* fenc_w2t = (__bf16*)carve(p, (size_t)128*128*2);
  __bf16* preWt  = (__bf16*)carve(p, (size_t)128*128*2);
  __bf16* t1g    = (__bf16*)carve(p, (size_t)NND*2);
  float*  basesF = (float*) carve(p, (size_t)NND*4);
  __bf16* edgebf = (__bf16*)carve(p, (size_t)NND*2);

  // --- weights / inputs staging ---
  k_prep_w<<<128, 256, 0, stream>>>(fe_w1, fe_w2, fenc_w2, c0_pre_w,
                                    fe_w1t, fe_w2t, fenc_w2t, preWt);
  k_concat<<<512, 256, 0, stream>>>(node, hidden, zbf);
  k_cvt<<<NND / 256, 256, 0, stream>>>(edge, edgebf, (long)NND);

  // --- spectral decomposition ---
  k_lap<<<2, 256, 0, stream>>>(adj, lap);
  hipFuncSetAttribute((const void*)k_jacobi,
                      hipFuncAttributeMaxDynamicSharedMemorySize, 262144);
  k_jacobi<<<Bn, 256, 262144, stream>>>(lap, V, evals);
  k_specprep<<<512, 256, 0, stream>>>(V, evals, dec_w, dec_b, Abf, Ebf);

  // --- node feature encoder: h0 = (relu(z@fe_w1+b1))@fe_w2 + b2 ---
  k_gemm<<<32, 256, 0, stream>>>(zbf, fe_w1t, fe_b1, u, Bn*Nn, Dn, 2*INn, 1);
  k_cvt<<<256, 256, 0, stream>>>(u, ubf, (long)Bn*Nn*Dn);
  k_gemm<<<32, 256, 0, stream>>>(ubf, fe_w2t, fe_b2, h0, Bn*Nn, Dn, Dn, 0);

  // --- spectral filters (16 GEMMs of 256^3, WMMA) ---
  k_filters<<<512, 256, 0, stream>>>(Abf, Ebf, filt);

  // --- filter encoder ---
  k_bases1<<<ROWS, 128, 0, stream>>>(filt, fenc_w1, fenc_b1,
                                     fenc_ln1_g, fenc_ln1_b, t1g);
  k_bases2<<<ROWS / 16, 256, 0, stream>>>(t1g, fenc_w2t, fenc_b2,
                                          fenc_ln2_g, fenc_ln2_b, adj, basesF);

  // --- SpecFormerConv (NL=1): e = gelu(edge@pre_w+b)*bases ---
  k_conv<<<ROWS / 16, 256, 0, stream>>>(edgebf, preWt, c0_pre_b, basesF, e_out);

  // --- h = h0 + e.sum(axis=1); residual MLP ---
  k_esum<<<256, 256, 0, stream>>>(e_out, h0, hbuf);
  k_final<<<Bn * Nn, 128, 0, stream>>>(hbuf, c0_w1, c0_b1, c0_ln1_g, c0_ln1_b,
                                       c0_w2, c0_b2, c0_ln2_g, c0_ln2_b, h_out);
}
